// CreateMultiAttentionGraphTransitions_23562190586026
// MI455X (gfx1250) — matrix operand order repaired and checked by reading, not compiled, not run on hardware
//
#include <hip/hip_runtime.h>

// ---------------------------------------------------------------------------
// MI455X (gfx1250) implementation: graph edge-MLP attention.
//   V=100000 vertices, F=P=64, K=16 neighbors, H=4 heads, hidden=128.
// All GEMMs run on v_wmma_f32_16x16x32_bf16 (wave32). Weights staged in LDS.
// Neighbor gather is software-pipelined; A-fragments packed with v_perm_b32.
// ---------------------------------------------------------------------------

typedef __attribute__((ext_vector_type(16))) __bf16          v16bf;
typedef __attribute__((ext_vector_type(8)))  float           v8f;
typedef __attribute__((ext_vector_type(8)))  unsigned short  u16x8;

#define VN     100000
#define NTILE  6250          // VN / 16
#define KNB    16
#define HID    128

union AFrag { v16bf v; u16x8 q[2]; unsigned short h[16]; unsigned int u[8]; };

// exact RNE scalar conversion (staging / scatter stores only)
__device__ __forceinline__ unsigned short f2bf(float f) {
  union { float f; unsigned int u; } c; c.f = f;
  unsigned int b = c.u;
  b += 0x7FFFu + ((b >> 16) & 1u);
  return (unsigned short)(b >> 16);
}
// paired bf16 pack: 2x v_add + 1x v_perm_b32 (round-to-nearest, ties away)
__device__ __forceinline__ unsigned int pack2bf(float lo, float hi) {
  union { float f; unsigned int u; } a, b;
  a.f = lo; b.f = hi;
  return __builtin_amdgcn_perm(b.u + 0x8000u, a.u + 0x8000u, 0x07060302u);
}
__device__ __forceinline__ float eluf(float x) {
  return x > 0.f ? x : (__expf(x) - 1.f);
}
__device__ __forceinline__ v8f wmma_bf16(v16bf a, v16bf b, v8f c) {
  return __builtin_amdgcn_wmma_f32_16x16x32_bf16(false, a, false, b, (short)0, c,
                                                 false, false);
}
__device__ __forceinline__ v16bf lds16(const unsigned short* p) {
  return *(const v16bf*)p;                  // 32B LDS read (2x ds_load_b128)
}

// ---------------------------------------------------------------------------
// Stage 1: xp = elu(x @ W_pre + b_pre)   [V,64] -> [V,64] f32 (workspace)
// ---------------------------------------------------------------------------
__global__ __launch_bounds__(256, 1) void stage1_pre(const float* __restrict__ x,
                                                     const float* __restrict__ Wp,
                                                     const float* __restrict__ bp,
                                                     float* __restrict__ xp) {
  __shared__ unsigned short sW[64 * 64];    // bf16, col-major [n*64 + k]
  __shared__ float sb[64];
  const int tid = threadIdx.x;
  for (int i = tid; i < 64 * 64; i += 256) {
    const int n = i & 63, k = i >> 6;
    sW[n * 64 + k] = f2bf(Wp[k * 64 + n]);
  }
  if (tid < 64) sb[tid] = bp[tid];
  __syncthreads();

  const int wave = tid >> 5, lane = tid & 31;
  const int row = lane & 15, grp = lane >> 4;
  const int tile = blockIdx.x * 8 + wave;
  if (tile >= NTILE) return;

  const float* xr = x + (size_t)(tile * 16 + row) * 64;
  float xv[32];
#pragma unroll
  for (int b = 0; b < 4; b++) {
    const int f0 = (b & 1) * 8 + (b >> 1) * 32 + grp * 16;
    const float4 p0 = *(const float4*)(xr + f0);
    const float4 p1 = *(const float4*)(xr + f0 + 4);
    xv[b * 8 + 0] = p0.x; xv[b * 8 + 1] = p0.y; xv[b * 8 + 2] = p0.z; xv[b * 8 + 3] = p0.w;
    xv[b * 8 + 4] = p1.x; xv[b * 8 + 5] = p1.y; xv[b * 8 + 6] = p1.z; xv[b * 8 + 7] = p1.w;
  }
  AFrag a0, a1;
#pragma unroll
  for (int i = 0; i < 8; i++) {
    a0.u[i] = pack2bf(xv[2 * i], xv[2 * i + 1]);
    a1.u[i] = pack2bf(xv[16 + 2 * i], xv[16 + 2 * i + 1]);
  }

  const v8f vzero = {0.f, 0.f, 0.f, 0.f, 0.f, 0.f, 0.f, 0.f};
#pragma unroll
  for (int nt = 0; nt < 4; nt++) {
    v8f acc = vzero;
    const int coff = (nt * 16 + row) * 64;
    acc = wmma_bf16(a0.v, lds16(sW + coff + grp * 16), acc);
    acc = wmma_bf16(a1.v, lds16(sW + coff + 32 + grp * 16), acc);
    const int n = nt * 16 + row;
    const float bn = sb[n];
#pragma unroll
    for (int j = 0; j < 8; j++)
      xp[(size_t)(tile * 16 + grp * 8 + j) * 64 + n] = eluf(acc[j] + bn);
  }
}

// ---------------------------------------------------------------------------
// Stage 2: per 16-vertex tile:
//   h0 = elu( sum_k (xp_v - xp_nk) @ W0_k  +  distsq @ W0_d  + b0 )
//   h1 = elu( h0 @ W1 + b1 );  logits = h1 @ W2 + b2;  per-head softmax(K=16)
// ---------------------------------------------------------------------------
__global__ __launch_bounds__(256, 1) void stage2_mlp(
    const float* __restrict__ xp, const int* __restrict__ nidx,
    const float* __restrict__ dsq,
    const float* __restrict__ W0, const float* __restrict__ b0,
    const float* __restrict__ W1, const float* __restrict__ b1,
    const float* __restrict__ W2, const float* __restrict__ b2,
    float* __restrict__ out) {
  extern __shared__ unsigned short smem[];
  unsigned short* sW0  = smem;               // 16*128*64 = 131072 halves (256 KB)
  unsigned short* sW0d = sW0 + 131072;       // 128*32 (K padded to 32)  (8 KB)
  unsigned short* sW1  = sW0d + 4096;        // 128*128 col-major       (32 KB)
  unsigned short* sW2  = sW1 + 16384;        // 64*128 col-major        (16 KB)
  float* sB0 = (float*)(sW2 + 8192);         // 128
  float* sB1 = sB0 + 128;                    // 128
  float* sB2 = sB1 + 128;                    // 64

  const int tid = threadIdx.x;

  // ---- cooperative weight staging (f32 global -> bf16 LDS, B-layouts) ----
  for (int it = tid; it < 16 * 64 * 32; it += 256) {  // W0 diff rows
    const int nq = it & 31;
    const int kr = it >> 5;
    const int r = kr & 63, k = kr >> 6;
    const float4 g = *(const float4*)(W0 + (size_t)(k * 65 + r) * 128 + nq * 4);
    unsigned short* d = sW0 + (size_t)(k * 128 + nq * 4) * 64 + r;
    d[0 * 64] = f2bf(g.x); d[1 * 64] = f2bf(g.y);
    d[2 * 64] = f2bf(g.z); d[3 * 64] = f2bf(g.w);
  }
  for (int i = tid; i < 128 * 32; i += 256) {         // W0 dist rows, K-padded
    const int k = i & 31, n = i >> 5;
    sW0d[n * 32 + k] =
        (!(k & 8)) ? f2bf(W0[(size_t)(((k & 7) + ((k >> 4) << 3)) * 65 + 64) * 128 + n]) : (unsigned short)0;
  }
  for (int i = tid; i < 128 * 128; i += 256) {        // W1 col-major
    const int k = i & 127, n = i >> 7;
    sW1[n * 128 + k] = f2bf(W1[(size_t)k * 128 + n]);
  }
  for (int i = tid; i < 64 * 128; i += 256) {         // W2 col-major
    const int k = i & 127, n = i >> 7;
    sW2[n * 128 + k] = f2bf(W2[(size_t)k * 64 + n]);
  }
  if (tid < 128) sB0[tid] = b0[tid];
  else if (tid < 256) sB1[tid - 128] = b1[tid - 128];
  if (tid < 64) sB2[tid] = b2[tid];
  __syncthreads();

  const int wave = tid >> 5, lane = tid & 31;
  const int row = lane & 15, grp = lane >> 4;
  const int tile = blockIdx.x * 8 + wave;
  const bool active = tile < NTILE;           // wave-uniform -> EXEC stays full

  const v8f vzero = {0.f, 0.f, 0.f, 0.f, 0.f, 0.f, 0.f, 0.f};
  v8f acc[8];
  if (active) {
#pragma unroll
    for (int nt = 0; nt < 8; nt++) acc[nt] = vzero;

    // center-vertex features for this lane's A-fragment blocks
    const float* xr = xp + (size_t)(tile * 16 + row) * 64;
    float xv[32];
#pragma unroll
    for (int b = 0; b < 4; b++) {
      const int f0 = (b & 1) * 8 + (b >> 1) * 32 + grp * 16;
      const float4 p0 = *(const float4*)(xr + f0);
      const float4 p1 = *(const float4*)(xr + f0 + 4);
      xv[b * 8 + 0] = p0.x; xv[b * 8 + 1] = p0.y; xv[b * 8 + 2] = p0.z; xv[b * 8 + 3] = p0.w;
      xv[b * 8 + 4] = p1.x; xv[b * 8 + 5] = p1.y; xv[b * 8 + 6] = p1.z; xv[b * 8 + 7] = p1.w;
    }

    // preload all 16 neighbor indices for this row (64B contiguous)
    int idxs[16];
    {
      const int* nrow = nidx + (size_t)(tile * 16 + row) * 16;
#pragma unroll
      for (int q = 0; q < 4; q++) {
        const int4 iv = *(const int4*)(nrow + q * 4);
        idxs[q * 4 + 0] = iv.x; idxs[q * 4 + 1] = iv.y;
        idxs[q * 4 + 2] = iv.z; idxs[q * 4 + 3] = iv.w;
      }
    }

    // distsq rank-block GEMM: [16 x 16(pad32)] x [32 x 128]
    {
      AFrag ad;
      const float* dr = dsq + (size_t)(tile * 16 + row) * 16 + grp * 8;
      const float4 d0 = *(const float4*)(dr);
      const float4 d1 = *(const float4*)(dr + 4);
      ad.u[0] = pack2bf(d0.x, d0.y); ad.u[1] = pack2bf(d0.z, d0.w);
      ad.u[2] = pack2bf(d1.x, d1.y); ad.u[3] = pack2bf(d1.z, d1.w);
      ad.u[4] = 0; ad.u[5] = 0; ad.u[6] = 0; ad.u[7] = 0;
#pragma unroll
      for (int nt = 0; nt < 8; nt++)
        acc[nt] = wmma_bf16(ad.v, lds16(sW0d + (nt * 16 + row) * 32 + grp * 16), acc[nt]);
    }

    // ---- software-pipelined neighbor loop: 16 x GEMM [16x64]x[64x128] ----
    float4 nb[8], nb2[8];
    {
      const int gi0 = idxs[0] >= 0 ? idxs[0] : 0;
      const float* nr = xp + (size_t)gi0 * 64;
#pragma unroll
      for (int b = 0; b < 4; b++) {
        const int f0 = (b & 1) * 8 + (b >> 1) * 32 + grp * 16;
        nb[2 * b + 0] = *(const float4*)(nr + f0);
        nb[2 * b + 1] = *(const float4*)(nr + f0 + 4);
      }
    }
#pragma unroll 1
    for (int k = 0; k < KNB; k++) {
      if (k < KNB - 1) {  // issue gathers for neighbor k+1 (hide L2 latency)
        const int gin = idxs[k + 1] >= 0 ? idxs[k + 1] : 0;
        const float* nr = xp + (size_t)gin * 64;
#pragma unroll
        for (int b = 0; b < 4; b++) {
          const int f0 = (b & 1) * 8 + (b >> 1) * 32 + grp * 16;
          nb2[2 * b + 0] = *(const float4*)(nr + f0);
          nb2[2 * b + 1] = *(const float4*)(nr + f0 + 4);
        }
      }
      const float msk = idxs[k] >= 0 ? 1.f : 0.f;
      AFrag a0, a1;
#pragma unroll
      for (int b = 0; b < 4; b++) {
        const float4 p0 = nb[2 * b + 0];
        const float4 p1 = nb[2 * b + 1];
        const float d0 = xv[b * 8 + 0] - msk * p0.x;
        const float d1 = xv[b * 8 + 1] - msk * p0.y;
        const float d2 = xv[b * 8 + 2] - msk * p0.z;
        const float d3 = xv[b * 8 + 3] - msk * p0.w;
        const float d4 = xv[b * 8 + 4] - msk * p1.x;
        const float d5 = xv[b * 8 + 5] - msk * p1.y;
        const float d6 = xv[b * 8 + 6] - msk * p1.z;
        const float d7 = xv[b * 8 + 7] - msk * p1.w;
        AFrag* af = (b < 2) ? &a0 : &a1;
        const int ub = (b & 1) * 4;
        af->u[ub + 0] = pack2bf(d0, d1);
        af->u[ub + 1] = pack2bf(d2, d3);
        af->u[ub + 2] = pack2bf(d4, d5);
        af->u[ub + 3] = pack2bf(d6, d7);
      }
      const unsigned short* wb = sW0 + (size_t)k * 128 * 64;
#pragma unroll
      for (int nt = 0; nt < 8; nt++) {
        const int coff = (nt * 16 + row) * 64;
        acc[nt] = wmma_bf16(a0.v, lds16(wb + coff + grp * 16), acc[nt]);
        acc[nt] = wmma_bf16(a1.v, lds16(wb + coff + 32 + grp * 16), acc[nt]);
      }
#pragma unroll
      for (int i = 0; i < 8; i++) nb[i] = nb2[i];
    }
    // bias + ELU (h0 kept in registers across the barrier)
#pragma unroll
    for (int nt = 0; nt < 8; nt++) {
      const float bn = sB0[nt * 16 + row];
#pragma unroll
      for (int j = 0; j < 8; j++) acc[nt][j] = eluf(acc[nt][j] + bn);
    }
  }
  __syncthreads();  // everyone done reading sW0 -> safe to reuse as scratch

  unsigned short* hs0 = sW0 + wave * 2048;            // h0 bf16 [16][128]
  unsigned short* hs1 = sW0 + 16384 + wave * 2048;    // h1 bf16 [16][128]
  float* sf = (float*)(sW0 + 32768) + wave * 1024;    // logits f32 [16][64]

  if (active) {
    // C-layout -> row-major LDS (re-fragment for next GEMM's A operand)
#pragma unroll
    for (int nt = 0; nt < 8; nt++)
#pragma unroll
      for (int j = 0; j < 8; j++)
        hs0[(grp * 8 + j) * 128 + nt * 16 + row] = f2bf(acc[nt][j]);
    asm volatile("s_wait_dscnt 0" ::: "memory");

    // GEMM2: h0 @ W1  ([16x128] x [128x128])
    v8f acc2[8];
#pragma unroll
    for (int nt = 0; nt < 8; nt++) acc2[nt] = vzero;
#pragma unroll
    for (int kb = 0; kb < 4; kb++) {
      AFrag a;
      a.q[0] = *(const u16x8*)(hs0 + row * 128 + kb * 32 + grp * 16);
      a.q[1] = *(const u16x8*)(hs0 + row * 128 + kb * 32 + 8 + grp * 16);
#pragma unroll
      for (int nt = 0; nt < 8; nt++)
        acc2[nt] = wmma_bf16(a.v, lds16(sW1 + (nt * 16 + row) * 128 + kb * 32 + grp * 16),
                             acc2[nt]);
    }
#pragma unroll
    for (int nt = 0; nt < 8; nt++) {
      const float bn = sB1[nt * 16 + row];
#pragma unroll
      for (int j = 0; j < 8; j++) acc2[nt][j] = eluf(acc2[nt][j] + bn);
    }
#pragma unroll
    for (int nt = 0; nt < 8; nt++)
#pragma unroll
      for (int j = 0; j < 8; j++)
        hs1[(grp * 8 + j) * 128 + nt * 16 + row] = f2bf(acc2[nt][j]);
    asm volatile("s_wait_dscnt 0" ::: "memory");

    // GEMM3: h1 @ W2  ([16x128] x [128x64])
    v8f acc3[4];
#pragma unroll
    for (int nt = 0; nt < 4; nt++) acc3[nt] = vzero;
#pragma unroll
    for (int kb = 0; kb < 4; kb++) {
      AFrag a;
      a.q[0] = *(const u16x8*)(hs1 + row * 128 + kb * 32 + grp * 16);
      a.q[1] = *(const u16x8*)(hs1 + row * 128 + kb * 32 + 8 + grp * 16);
#pragma unroll
      for (int nt = 0; nt < 4; nt++)
        acc3[nt] = wmma_bf16(a.v, lds16(sW2 + (nt * 16 + row) * 128 + kb * 32 + grp * 16),
                             acc3[nt]);
    }
    // logits -> LDS f32
#pragma unroll
    for (int nt = 0; nt < 4; nt++) {
      const float bn = sB2[nt * 16 + row];
#pragma unroll
      for (int j = 0; j < 8; j++)
        sf[(grp * 8 + j) * 64 + nt * 16 + row] = acc3[nt][j] + bn;
    }
    asm volatile("s_wait_dscnt 0" ::: "memory");

    // per-head softmax over K=16; 64 groups per tile -> 2 per lane
#pragma unroll
    for (int g2 = 0; g2 < 2; g2++) {
      const int g = lane * 2 + g2;
      const int r = g >> 2, head = g & 3;
      const float* src = sf + r * 64 + head * 16;
      float m = src[0];
#pragma unroll
      for (int i = 1; i < 16; i++) m = fmaxf(m, src[i]);
      float e[16], s = 0.f;
#pragma unroll
      for (int i = 0; i < 16; i++) { e[i] = __expf(src[i] - m); s += e[i]; }
      const float inv = 1.f / s;
      float* dst = out + (size_t)(tile * 16 + r) * 64 + head * 16;
#pragma unroll
      for (int q = 0; q < 4; q++) {
        float4 o;
        o.x = e[q * 4 + 0] * inv; o.y = e[q * 4 + 1] * inv;
        o.z = e[q * 4 + 2] * inv; o.w = e[q * 4 + 3] * inv;
        *(float4*)(dst + q * 4) = o;
      }
    }
  }
}

// ---------------------------------------------------------------------------
extern "C" void kernel_launch(void* const* d_in, const int* in_sizes, int n_in,
                              void* d_out, int out_size, void* d_ws, size_t ws_size,
                              hipStream_t stream) {
  (void)in_sizes; (void)n_in; (void)out_size; (void)ws_size;
  const float* x    = (const float*)d_in[0];
  const int*   nidx = (const int*)d_in[1];
  const float* dsq  = (const float*)d_in[2];
  const float* Wp   = (const float*)d_in[3];
  const float* bp   = (const float*)d_in[4];
  const float* W0   = (const float*)d_in[5];
  const float* b0   = (const float*)d_in[6];
  const float* W1   = (const float*)d_in[7];
  const float* b1   = (const float*)d_in[8];
  const float* W2   = (const float*)d_in[9];
  const float* b2   = (const float*)d_in[10];
  float* out = (float*)d_out;
  float* xp  = (float*)d_ws;  // [V,64] f32 = 25.6 MB (L2-resident)

  const int blocks = (NTILE + 7) / 8;  // 782
  stage1_pre<<<blocks, 256, 0, stream>>>(x, Wp, bp, xp);

  const size_t lds2 =
      (size_t)(131072 + 4096 + 16384 + 8192) * sizeof(unsigned short) +
      (size_t)(128 + 128 + 64) * sizeof(float);  // ~313 KB (<= 320 KB/WGP)
  stage2_mlp<<<blocks, 256, lds2, stream>>>(xp, nidx, dsq, W0, b0, W1, b1, W2, b2, out);
}
